// VolterraLayer2D_59554016526696
// MI455X (gfx1250) — matrix operand order repaired and checked by reading, not compiled, not run on hardware
//
#include <hip/hip_runtime.h>

// MI455X / gfx1250 implementation of the Volterra layer.
//
// Folding: rank-4 two-layer 1x1 projections per offset pair collapse to one
// 64x64 matrix E_p = sum_q Wb[p,q] @ Wa[p,q] (pre-scaled by 1/81); the 3x3
// conv is 9 more 64x64 matrices.  Total: 90 GEMM terms of
// (64x64) x (64 x B*H*W) ~= 48.3 GFLOP over ~4 MB of activations ->
// compute-bound -> v_wmma_f32_16x16x32_bf16 with f32 accumulation.
//
// A-matrices are pre-packed in the exact WMMA A-fragment lane/VGPR layout
// (720 KB, L2-resident across all 512 blocks) and register-double-buffered
// one K-step ahead so global load latency hides behind the WMMAs.
// x is pre-transposed to bf16 (w,c)-major so the per-block LDS staging is a
// contiguous 48 KB copy via global_load_async_to_lds_b128 (ASYNCcnt path).

#define BB    4
#define CC    64
#define HH    128
#define WW    128
#define NOFF  9
#define NPAIR 81
#define RANK  4
#define NTERM 90   // 81 pair terms + 9 conv taps
#define NU    (NTERM * 2)

typedef __attribute__((ext_vector_type(8)))  float   v8f;
typedef __attribute__((ext_vector_type(8)))  __bf16  v8bf;
typedef __attribute__((ext_vector_type(16))) __bf16  v16bf;

// Exact parameter type of the async-load builtin (from the compiler
// diagnostic): gcc-style int4 vector, global (AS1) source / LDS (AS3) dest.
typedef int v4i_vs __attribute__((vector_size(16)));
typedef __attribute__((address_space(1))) v4i_vs* v4i_gp;
typedef __attribute__((address_space(3))) v4i_vs* v4i_lp;

#if __has_builtin(__builtin_amdgcn_global_load_async_to_lds_b128) && \
    __has_builtin(__builtin_amdgcn_s_wait_asynccnt)
#define USE_ASYNC_LDS 1
#else
#define USE_ASYNC_LDS 0
#endif

__device__ __forceinline__ unsigned short f2bf(float f) {
  unsigned u = __float_as_uint(f);
  unsigned r = u + 0x7FFFu + ((u >> 16) & 1u);   // round-to-nearest-even
  return (unsigned short)(r >> 16);
}

// Copy 16 bytes global -> LDS (async DMA path when the builtin exists).
// AS1 pointer: generic global address is numerically identical.
// AS3 pointer: low 32 bits of a generic LDS pointer are the LDS offset.
__device__ __forceinline__ void copy16_g2l(unsigned short* lds,
                                           const unsigned short* g) {
#if USE_ASYNC_LDS
  __builtin_amdgcn_global_load_async_to_lds_b128(
      (v4i_gp)(uintptr_t)g, (v4i_lp)(unsigned)(uintptr_t)lds, 0, 0);
#else
  *(uint4*)lds = *(const uint4*)g;
#endif
}

// Index into the packed A-fragment buffer for term t, element (po, c).
// Fragment = 32 lanes x 16 halves (1 KB) per (t, kstep, mtile).
// 16-bit A-matrix 16x32 layout used here (contiguous k, not ISA table order):
//   lanes 0-15 (g=0): V0..7 hold K=0..15
//   lanes16-31 (g=1): V0..7 hold K=16..31, M = lane%16
__device__ __forceinline__ int apack_index(int t, int po, int c) {
  int mt = po >> 4, m = po & 15;
  int ks = c >> 5,  kl = c & 31;
  int g   = kl >> 4;
  int klp = kl - (g << 4);
  int v   = klp >> 1;
  int hi  = klp & 1;
  return (((t * 2 + ks) * 4 + mt) << 9) | (((g << 4) + m) << 4) | ((v << 1) + hi);
}

// ---------------------------------------------------------------------------
// Kernel 1: fold Wa/Wb (pairs, scaled by 1/81) and W1 taps into packed bf16
// A-fragments.  One block per term.
// ---------------------------------------------------------------------------
__global__ __launch_bounds__(256) void volterra_fold_kernel(
    const float* __restrict__ Wa, const float* __restrict__ Wb,
    const float* __restrict__ W1, unsigned short* __restrict__ Apack) {
  int p = blockIdx.x;
  if (p < NPAIR) {
    __shared__ float WaSh[CC * CC];
    int po = threadIdx.x >> 2;          // 0..63 output channel
    int c0 = (threadIdx.x & 3) << 4;    // 16-wide input-channel slab
    float e[16];
#pragma unroll
    for (int k = 0; k < 16; ++k) e[k] = 0.0f;
    for (int q = 0; q < RANK; ++q) {
      __syncthreads();
      for (int idx = threadIdx.x; idx < CC * CC; idx += 256)
        WaSh[idx] = Wa[(p * RANK + q) * CC * CC + idx];
      __syncthreads();
      const float* wbrow = Wb + ((p * RANK + q) * CC + po) * CC;
      for (int o = 0; o < CC; ++o) {
        float wb = wbrow[o];
        const float* wa = &WaSh[o * CC + c0];
#pragma unroll
        for (int k = 0; k < 16; ++k) e[k] += wb * wa[k];
      }
    }
    const float inv = 1.0f / (float)NPAIR;
#pragma unroll
    for (int k = 0; k < 16; ++k)
      Apack[apack_index(p, po, c0 + k)] = f2bf(e[k] * inv);
  } else {
    int o = p - NPAIR;
    int ky = o / 3, kx = o % 3;
    for (int idx = threadIdx.x; idx < CC * CC; idx += 256) {
      int po = idx >> 6, c = idx & 63;
      float v = W1[((po * CC + c) * 3 + ky) * 3 + kx];
      Apack[apack_index(p, po, c)] = f2bf(v);
    }
  }
}

// ---------------------------------------------------------------------------
// Kernel 2: fold all biases into one per-channel vector.
// out_bias = b1 + (sum_{p,q} [ Wb[p,q] @ ba[p,q] + bb[p,q] ]) / 81
// ---------------------------------------------------------------------------
__global__ __launch_bounds__(64) void volterra_bias_kernel(
    const float* __restrict__ b1, const float* __restrict__ Wb,
    const float* __restrict__ ba, const float* __restrict__ bb,
    float* __restrict__ biasv) {
  int po = threadIdx.x;
  float acc = b1[po];
  const float inv = 1.0f / (float)NPAIR;
  for (int p = 0; p < NPAIR; ++p)
    for (int q = 0; q < RANK; ++q) {
      const float* wbrow = Wb + ((p * RANK + q) * CC + po) * CC;
      const float* barow = ba + (p * RANK + q) * CC;
      float s = 0.0f;
      for (int o = 0; o < CC; ++o) s += wbrow[o] * barow[o];
      acc += (s + bb[(p * RANK + q) * CC + po]) * inv;
    }
  biasv[po] = acc;
}

// ---------------------------------------------------------------------------
// Kernel 3: transpose x to bf16, (b, h, w, c)-major, so image rows are
// contiguous 16 KB slabs ready for straight global->LDS copies.
// ---------------------------------------------------------------------------
__global__ __launch_bounds__(256) void volterra_xpose_kernel(
    const float* __restrict__ x, unsigned short* __restrict__ xT) {
  int b = blockIdx.x >> 7;
  int row = blockIdx.x & (HH - 1);
  for (int idx = threadIdx.x; idx < CC * WW; idx += 256) {
    int c = idx >> 7, w = idx & 127;
    float v = x[((b * CC + c) * HH + row) * WW + w];
    xT[((size_t)(b * HH + row) * WW + w) * CC + c] = f2bf(v);
  }
}

// ---------------------------------------------------------------------------
// Main kernel: one block per (batch, output row).  8 waves x 16-pixel N-tiles.
// Per wave: 90 terms x 2 K-steps x 4 M-tiles of WMMA_F32_16X16X32_BF16, with
// the 4 A-fragments register-double-buffered one K-step ahead.
// ---------------------------------------------------------------------------
__global__ __launch_bounds__(256) void volterra_main_kernel(
    const unsigned short* __restrict__ xT,
    const unsigned short* __restrict__ Apack,
    const float* __restrict__ biasv, float* __restrict__ out) {
  __shared__ unsigned short xsh[3 * WW * CC];
  __shared__ float biasSh[CC];

  int b = blockIdx.x >> 7;
  int h = blockIdx.x & (HH - 1);
  int tid = threadIdx.x;

  // Stage rows h-1, h, h+1 (circular) as bf16: contiguous 16 KB per row slab.
  for (int idx = tid; idx < (3 * WW * CC) / 8; idx += 256) {  // b128 chunks
    int r = idx >> 10;           // 1024 chunks per slab
    int e = idx & 1023;
    int row = (h + r - 1 + HH) & (HH - 1);
    const unsigned short* g =
        xT + ((size_t)(b * HH + row) * WW) * CC + (size_t)e * 8;
    copy16_g2l(&xsh[r * WW * CC + e * 8], g);
  }
  if (tid < CC) biasSh[tid] = biasv[tid];
#if USE_ASYNC_LDS
  __builtin_amdgcn_s_wait_asynccnt(0);
#endif
  __syncthreads();

  int lane = tid & 31;
  int wv   = tid >> 5;       // wave id == N-tile (16 pixels each)
  int g    = lane >> 4;      // lane half-group
  int n    = lane & 15;
  int wout = wv * 16 + n;    // this lane's output pixel column

  v8f acc[4] = {};           // 4 M-tiles x 8 f32 accumulator VGPRs
  const v8bf zerov = {};

  // Preload A-fragments for u = 0 (term 0, kstep 0).
  const unsigned short* alane = Apack + lane * 16;
  v16bf curA[4];
#pragma unroll
  for (int mt = 0; mt < 4; ++mt)
    curA[mt] = *reinterpret_cast<const v16bf*>(alane + mt * 512);

  for (int t = 0; t < NTERM; ++t) {
    if (t + 2 < NTERM)
      __builtin_prefetch(Apack + (t + 2) * 4096, 0, 3);  // global_prefetch_b8

    bool isConv = (t >= NPAIR);
    int rowA, colA, rowB = 0, colB = 0;
    bool okA = true;
    if (!isConv) {
      // Xs[o][h,w] = x[(h-dy)%H, (w-dx)%W]; local rows: 0=h-1,1=h,2=h+1.
      int i = t / NOFF, j = t % NOFF;
      int dyi = i / 3 - 1, dxi = i % 3 - 1;
      int dyj = j / 3 - 1, dxj = j % 3 - 1;
      rowA = 1 - dyi; colA = (wout - dxi) & (WW - 1);
      rowB = 1 - dyj; colB = (wout - dxj) & (WW - 1);
    } else {
      // Zero-padded conv tap: x[h+ky-1, w+kx-1].
      int o = t - NPAIR;
      int ky = o / 3, kx = o % 3;
      rowA = ky;
      colA = wout + kx - 1;
      int hy = h + ky - 1;
      okA = (colA >= 0) & (colA < WW) & (hy >= 0) & (hy < HH);
      if (!okA) colA = 0;
    }

#pragma unroll
    for (int ks = 0; ks < 2; ++ks) {
      int u  = t * 2 + ks;
      int un = (u + 1 < NU) ? u + 1 : u;   // clamp final iteration

      // Issue next K-step's A-fragment loads before consuming the current
      // ones, so s_wait_loadcnt covers a full iteration of latency.
      const unsigned short* nbase = Apack + (size_t)un * 2048 + lane * 16;
      v16bf nxtA[4];
#pragma unroll
      for (int mt = 0; mt < 4; ++mt)
        nxtA[mt] = *reinterpret_cast<const v16bf*>(nbase + mt * 512);

      // ---- build B fragment: prod[c, wout] for K = ks*32 + g*16 + s ----
      int c0 = ks * 32 + g * 16;
      const v8bf* pa =
          reinterpret_cast<const v8bf*>(&xsh[(rowA * WW + colA) * CC + c0]);
      v8bf alo = pa[0], ahi = pa[1];
      if (!okA) { alo = zerov; ahi = zerov; }
      if (!isConv) {
        const v8bf* pb =
            reinterpret_cast<const v8bf*>(&xsh[(rowB * WW + colB) * CC + c0]);
        alo = alo * pb[0];   // v_pk_mul_bf16
        ahi = ahi * pb[1];
      }
      v16bf bfrag = __builtin_shufflevector(alo, ahi, 0, 1, 2, 3, 4, 5, 6, 7,
                                            8, 9, 10, 11, 12, 13, 14, 15);

#pragma unroll
      for (int mt = 0; mt < 4; ++mt)
        acc[mt] = __builtin_amdgcn_wmma_f32_16x16x32_bf16(
            false, curA[mt], false, bfrag, (short)0, acc[mt], false, false);

#pragma unroll
      for (int mt = 0; mt < 4; ++mt) curA[mt] = nxtA[mt];
    }
  }

  // D layout: VGPR r -> M = r + 8*g (per M-tile); N = lane%16 = wout.
#pragma unroll
  for (int mt = 0; mt < 4; ++mt)
#pragma unroll
    for (int r = 0; r < 8; ++r) {
      int po = mt * 16 + r + g * 8;
      out[((b * CC + po) * HH + h) * WW + wout] = acc[mt][r] + biasSh[po];
    }
}

// ---------------------------------------------------------------------------
extern "C" void kernel_launch(void* const* d_in, const int* in_sizes, int n_in,
                              void* d_out, int out_size, void* d_ws,
                              size_t ws_size, hipStream_t stream) {
  const float* x  = (const float*)d_in[0];
  const float* W1 = (const float*)d_in[1];
  const float* b1 = (const float*)d_in[2];
  const float* Wa = (const float*)d_in[3];
  const float* ba = (const float*)d_in[4];
  const float* Wb = (const float*)d_in[5];
  const float* bb = (const float*)d_in[6];
  float* out = (float*)d_out;

  // Workspace layout:
  //   [0, 720 KB)      packed bf16 A-fragments (90 terms)
  //   [720 KB, +256 B) folded bias vector
  //   [+256 B, +8 MB)  bf16 transposed activations xT[b][h][w][c]
  unsigned short* Apack = (unsigned short*)d_ws;
  size_t apack_bytes = (size_t)NTERM * 4096 * sizeof(unsigned short);
  float* biasv = (float*)((char*)d_ws + apack_bytes);
  unsigned short* xT =
      (unsigned short*)((char*)d_ws + apack_bytes + 256);

  volterra_fold_kernel<<<NTERM, 256, 0, stream>>>(Wa, Wb, W1, Apack);
  volterra_bias_kernel<<<1, CC, 0, stream>>>(b1, Wb, ba, bb, biasv);
  volterra_xpose_kernel<<<BB * HH, 256, 0, stream>>>(x, xT);
  volterra_main_kernel<<<BB * HH, 256, 0, stream>>>(xT, Apack, biasv, out);
}
